// IMDB_model_25374666784999
// MI455X (gfx1250) — matrix operand order repaired and checked by reading, not compiled, not run on hardware
//
#include <hip/hip_runtime.h>

#define Bc 16
#define Sc 600
#define Ec 300
#define Hc 512
#define NCc 2
#define Dc (Sc*Ec)      // 180000
#define SP 608          // padded S (mult of 32)
#define EP 320          // padded E as K (mult of 32)
#define NP 304          // padded E as N (mult of 16)
#define KS 45           // K-split for g1 GEMM (180000/45 = 4000 per slice)
#define TSTEPS 3

typedef __bf16 bf16;
typedef __attribute__((ext_vector_type(16))) __bf16 v16bf;
typedef __attribute__((ext_vector_type(8)))  __bf16 v8bf;
typedef __attribute__((ext_vector_type(4)))  __bf16 v4bf;
typedef __attribute__((ext_vector_type(8)))  float  v8f;

__device__ __forceinline__ v16bf bf_cat(v8bf lo, v8bf hi){
  v16bf r;
#pragma unroll
  for (int i = 0; i < 8; ++i){ r[i] = lo[i]; r[i+8] = hi[i]; }
  return r;
}

// A operand (16x32 bf16): rows mb..mb+15 of row-major [ldk] bf16.
// Lane layout: lanes 0-15 hold K 0..15, lanes 16-31 hold K 16..31 (contiguous, matches loadBT).
__device__ __forceinline__ v16bf loadA(const bf16* base, int ldk, int mb, int kb, int lane){
  const int hf = lane >> 4;
  const int m  = mb + (lane & 15);
  const bf16* p = base + (size_t)m * ldk + kb + hf * 16;
  return bf_cat(*(const v8bf*)p, *(const v8bf*)(p + 8));
}

// B operand (32x16 bf16) from B stored transposed [n][k] row-major.
// Lane n reads 16 contiguous k: lanes 0-15 K 0..15, lanes 16-31 K 16..31.
__device__ __forceinline__ v16bf loadBT(const bf16* baseT, int ldk, int nb, int kb, int lane){
  const int hf = lane >> 4;
  const int n  = nb + (lane & 15);
  const bf16* p = baseT + (size_t)n * ldk + kb + hf * 16;
  return bf_cat(*(const v8bf*)p, *(const v8bf*)(p + 8));
}

__device__ __forceinline__ v8f wmma_bf(v16bf a, v16bf b, v8f c){
  return __builtin_amdgcn_wmma_f32_16x16x32_bf16(false, a, false, b, (short)0, c, false, false);
}

// ---------------- prep kernels ----------------

// W1 fp32 [512][180000] -> bf16 same layout (4 elems/thread, exact grid)
__global__ void k_cvt_w1(const float* __restrict__ W1, bf16* __restrict__ W1b){
  size_t i = ((size_t)blockIdx.x * blockDim.x + threadIdx.x) * 4;
  if (i >= (size_t)Hc * Dc) return;
  float4 v = *(const float4*)(W1 + i);
  v4bf o; o[0] = (bf16)v.x; o[1] = (bf16)v.y; o[2] = (bf16)v.z; o[3] = (bf16)v.w;
  *(v4bf*)(W1b + i) = o;
}

// x -> xb [B][SP][EP] (zero padded) and xTb [B][EP][SP]
__global__ void k_prep_x(const float* __restrict__ x, bf16* __restrict__ xb, bf16* __restrict__ xTb){
  int i = blockIdx.x * blockDim.x + threadIdx.x;       // B*SP*EP = 3,112,960
  if (i >= Bc * SP * EP) return;
  int e = i % EP; int s = (i / EP) % SP; int b = i / (EP * SP);
  float v = (s < Sc && e < Ec) ? x[((size_t)b * Sc + s) * Ec + e] : 0.f;
  bf16 h = (bf16)v;
  xb[i] = h;
  xTb[((size_t)b * EP + e) * SP + s] = h;
}

// W0 [300][300] -> W0b [NP][EP] zero padded (rows f, cols e_in)
__global__ void k_prep_w0(const float* __restrict__ W0, bf16* __restrict__ W0b){
  int i = blockIdx.x * blockDim.x + threadIdx.x;       // NP*EP = 97280
  if (i >= NP * EP) return;
  int e = i % EP, f = i / EP;
  W0b[i] = (bf16)((f < Ec && e < Ec) ? W0[f * Ec + e] : 0.f);
}

__global__ void k_zero8(bf16* __restrict__ p){
  size_t i = ((size_t)blockIdx.x * blockDim.x + threadIdx.x) * 8;
  v8bf z = {};
  *(v8bf*)(p + i) = z;
}

__global__ void k_init_n0(const float* __restrict__ n0i, float* __restrict__ n0o, bf16* __restrict__ n0b){
  int i = blockIdx.x * blockDim.x + threadIdx.x;       // B*D = 2,880,000
  if (i >= Bc * Dc) return;
  float v = n0i[i]; n0o[i] = v; n0b[i] = (bf16)v;
}

__global__ void k_init_n12(const float* __restrict__ n1i, const float* __restrict__ n2i,
                           float* __restrict__ n1o, float* __restrict__ n2o, bf16* __restrict__ n1b){
  int i = blockIdx.x * blockDim.x + threadIdx.x;       // 8192
  if (i < Bc * Hc){ float v = n1i[i]; n1o[i] = v; n1b[i] = (bf16)v; }
  if (i < Bc * NCc){ n2o[i] = n2i[i]; }
}

// ---------------- GEMM kernels ----------------

// Wx = x @ W0^T : M = B*SP (rows b*SP+s), N = NP, K = EP. Wave per 16x16 tile.
__global__ void k_wx(const bf16* __restrict__ xb, const bf16* __restrict__ W0b, float* __restrict__ Wx){
  int wave = blockIdx.x * (blockDim.x >> 5) + (threadIdx.x >> 5);
  int lane = threadIdx.x & 31;
  const int NT = NP / 16;                 // 19
  int mt = wave / NT, nt = wave % NT;     // mt < 608
  int mb = mt * 16, nb = nt * 16;
  v8f acc = {};
  for (int kb = 0; kb < EP; kb += 32){
    v16bf a = loadA(xb, EP, mb, kb, lane);
    v16bf b = loadBT(W0b, EP, nb, kb, lane);
    acc = wmma_bf(a, b, acc);
  }
  int hf = lane >> 4, nl = lane & 15;
#pragma unroll
  for (int r = 0; r < 8; ++r){
    int m = mb + r + hf * 8;
    Wx[(size_t)m * NP + nb + nl] = acc[r];
  }
}

// g0 = Wx + n1 @ W1 : M=16 (batch), N=D, K=H. Block = 5 waves covering 80 d,
// W1 k-chunk staged through LDS (coalesced global reads).
__global__ __launch_bounds__(160) void k_g0(const bf16* __restrict__ n1b, const bf16* __restrict__ W1b,
                                            const float* __restrict__ Wx, bf16* __restrict__ g0b){
  __shared__ __align__(16) bf16 lds[32 * 80];
  int tid = threadIdx.x, wave = tid >> 5, lane = tid & 31;
  int d0 = blockIdx.x * 80;               // grid = 2250, exact
  v8f acc = {};
  for (int kb = 0; kb < Hc; kb += 32){
    { // stage W1[kb..kb+31][d0..d0+79] -> lds[k][d]; 16 bf16 per thread
      int k = tid / 5, c = (tid % 5) * 16;
      const bf16* src = W1b + (size_t)(kb + k) * Dc + d0 + c;
      v8bf a0 = *(const v8bf*)src;
      v8bf a1 = *(const v8bf*)(src + 8);
      *(v8bf*)(lds + k * 80 + c)     = a0;
      *(v8bf*)(lds + k * 80 + c + 8) = a1;
    }
    __syncthreads();
    v16bf a = loadA(n1b, Hc, 0, kb, lane);
    v16bf b;
    {
      int hf = lane >> 4, n = wave * 16 + (lane & 15);
#pragma unroll
      for (int i = 0; i < 16; ++i) b[i] = lds[(hf * 16 + i) * 80 + n];
    }
    acc = wmma_bf(a, b, acc);
    __syncthreads();
  }
  int hf = lane >> 4, nl = lane & 15;
  int d = d0 + wave * 16 + nl;
  int s = d / Ec, e = d % Ec;
#pragma unroll
  for (int r = 0; r < 8; ++r){
    int m = r + hf * 8;                   // batch
    float v = acc[r] + Wx[((size_t)m * SP + s) * NP + e];
    g0b[((size_t)m * SP + s) * EP + e] = (bf16)v;
  }
}

// Fused attention: scores = scale * g0 @ x^T -> softmax rows -> attnb (bf16)
__global__ __launch_bounds__(256) void k_attn(const bf16* __restrict__ g0b, const bf16* __restrict__ xb,
                                              bf16* __restrict__ attnb){
  __shared__ float sc[16][SP];            // 38,912 B
  int b = blockIdx.y, mt = blockIdx.x;    // (38, 16)
  int tid = threadIdx.x, wave = tid >> 5, lane = tid & 31;
  const bf16* A = g0b + (size_t)b * SP * EP;
  const bf16* X = xb  + (size_t)b * SP * EP;
  int mb = mt * 16;
  const float scale = 0.05773502691896258f;   // 1/sqrt(300)
  for (int nt = wave; nt < SP / 16; nt += 8){
    v8f acc = {};
    for (int kb = 0; kb < EP; kb += 32){
      v16bf a  = loadA(A, EP, mb, kb, lane);
      v16bf bb = loadBT(X, EP, nt * 16, kb, lane);   // row t of x, contiguous e
      acc = wmma_bf(a, bb, acc);
    }
    int hf = lane >> 4, nl = lane & 15;
#pragma unroll
    for (int r = 0; r < 8; ++r) sc[r + hf * 8][nt * 16 + nl] = scale * acc[r];
  }
  __syncthreads();
  // softmax: each wave owns 2 rows
  for (int rr = 0; rr < 2; ++rr){
    int m = wave * 2 + rr;
    float mx = -1e30f;
    for (int t = lane; t < Sc; t += 32) mx = fmaxf(mx, sc[m][t]);
#pragma unroll
    for (int o = 16; o; o >>= 1) mx = fmaxf(mx, __shfl_xor(mx, o, 32));
    float sum = 0.f;
    for (int t = lane; t < Sc; t += 32){ float e = __expf(sc[m][t] - mx); sc[m][t] = e; sum += e; }
#pragma unroll
    for (int o = 16; o; o >>= 1) sum += __shfl_xor(sum, o, 32);
    float inv = 1.f / sum;
    bf16* out = attnb + ((size_t)b * SP + mb + m) * SP;
    for (int t = lane; t < Sc; t += 32) out[t] = (bf16)(sc[m][t] * inv);
    if (lane < SP - Sc) out[Sc + lane] = (bf16)0.f;   // zero padded columns
  }
}

// n0 = attn @ x : per batch M=SP, N=NP, K=SP. Writes fp32 n0 (d_out) + bf16 n0b.
__global__ void k_n0(const bf16* __restrict__ attnb, const bf16* __restrict__ xTb,
                     float* __restrict__ n0o, bf16* __restrict__ n0b){
  int wave = blockIdx.x * (blockDim.x >> 5) + (threadIdx.x >> 5);
  int lane = threadIdx.x & 31;
  const int MT = SP / 16, NT = NP / 16;   // 38, 19
  int per_b = MT * NT;                    // 722
  int b = wave / per_b, rem = wave % per_b;
  int mt = rem / NT, nt = rem % NT;
  const bf16* A  = attnb + (size_t)b * SP * SP;
  const bf16* Bt = xTb   + (size_t)b * EP * SP;     // [e][t]
  v8f acc = {};
  for (int kb = 0; kb < SP; kb += 32){
    v16bf a  = loadA(A, SP, mt * 16, kb, lane);
    v16bf bb = loadBT(Bt, SP, nt * 16, kb, lane);
    acc = wmma_bf(a, bb, acc);
  }
  int hf = lane >> 4, nl = lane & 15;
  int e = nt * 16 + nl;
#pragma unroll
  for (int r = 0; r < 8; ++r){
    int s = mt * 16 + r + hf * 8;
    if (s < Sc && e < Ec){
      size_t idx = ((size_t)b * Sc + s) * Ec + e;
      float v = acc[r];
      n0o[idx] = v;
      n0b[idx] = (bf16)v;
    }
  }
}

// g1 partials: n0 @ W1^T : M=16, N=H, K=D split KS ways. B rows = W1 rows (contiguous).
__global__ void k_g1(const bf16* __restrict__ n0b, const bf16* __restrict__ W1b, float* __restrict__ gpart){
  int wave_in = threadIdx.x >> 5, lane = threadIdx.x & 31;
  int nt = (blockIdx.x & 3) * 8 + wave_in;    // 0..31 (N=512)
  int ks = blockIdx.x >> 2;                   // 0..KS-1
  const int kspan = Dc / KS;                  // 4000 (mult of 32)
  int k0 = ks * kspan;
  v8f acc = {};
  for (int kb = k0; kb < k0 + kspan; kb += 32){
    v16bf a  = loadA(n0b, Dc, 0, kb, lane);
    v16bf bb = loadBT(W1b, Dc, nt * 16, kb, lane);  // W1[h][d] rows
    acc = wmma_bf(a, bb, acc);
  }
  int hf = lane >> 4, nl = lane & 15;
#pragma unroll
  for (int r = 0; r < 8; ++r){
    int m = r + hf * 8;                       // batch
    gpart[((size_t)ks * Bc + m) * Hc + nt * 16 + nl] = acc[r];
  }
}

// g2 = n1_old @ W2^T + b2 ; n2_tmp = tanh(g2)  (32 outputs, one wave)
__global__ void k_g2(const float* __restrict__ n1old, const float* __restrict__ W2,
                     const float* __restrict__ b2, float* __restrict__ n2t){
  int i = threadIdx.x;
  if (i >= Bc * NCc) return;
  int b = i >> 1, c = i & 1;
  float s = 0.f;
  for (int h = 0; h < Hc; ++h) s += n1old[b * Hc + h] * W2[c * Hc + h];
  n2t[i] = tanhf(s + b2[c]);
}

// n1 = tanh(sum_k gpart + b1 + n2_old @ W2)
__global__ void k_g1red(const float* __restrict__ gpart, const float* __restrict__ b1,
                        const float* __restrict__ n2old, const float* __restrict__ W2,
                        float* __restrict__ n1o, bf16* __restrict__ n1b){
  int i = blockIdx.x * blockDim.x + threadIdx.x;   // 8192
  if (i >= Bc * Hc) return;
  int b = i / Hc, h = i % Hc;
  float s = b1[h] + n2old[b * NCc + 0] * W2[0 * Hc + h] + n2old[b * NCc + 1] * W2[1 * Hc + h];
  for (int k = 0; k < KS; ++k) s += gpart[((size_t)k * Bc + b) * Hc + h];
  float v = tanhf(s);
  n1o[i] = v; n1b[i] = (bf16)v;
}

__global__ void k_n2commit(const float* __restrict__ n2t, float* __restrict__ n2o){
  int i = threadIdx.x;
  if (i < Bc * NCc) n2o[i] = n2t[i];
}

// ---------------- host ----------------

extern "C" void kernel_launch(void* const* d_in, const int* in_sizes, int n_in,
                              void* d_out, int out_size, void* d_ws, size_t ws_size,
                              hipStream_t stream){
  (void)in_sizes; (void)n_in; (void)out_size; (void)ws_size;
  const float* x   = (const float*)d_in[0];
  const float* n0i = (const float*)d_in[2];
  const float* n1i = (const float*)d_in[3];
  const float* n2i = (const float*)d_in[4];
  const float* W0  = (const float*)d_in[5];
  const float* W1  = (const float*)d_in[6];
  const float* b1  = (const float*)d_in[7];
  const float* W2  = (const float*)d_in[8];
  const float* b2  = (const float*)d_in[9];

  float* n0o = (float*)d_out;
  float* n1o = n0o + (size_t)Bc * Dc;
  float* n2o = n1o + (size_t)Bc * Hc;

  char* ws = (char*)d_ws;
  size_t off = 0;
  auto take = [&](size_t bytes) -> char* {
    char* p = ws + off; off = (off + bytes + 255) & ~(size_t)255; return p;
  };
  bf16*  W1b   = (bf16*) take((size_t)Hc * Dc * 2);          // 184.3 MB (fits L2)
  bf16*  xb    = (bf16*) take((size_t)Bc * SP * EP * 2);
  bf16*  xTb   = (bf16*) take((size_t)Bc * EP * SP * 2);
  bf16*  W0b   = (bf16*) take((size_t)NP * EP * 2);
  float* Wx    = (float*)take((size_t)Bc * SP * NP * 4);
  bf16*  g0b   = (bf16*) take((size_t)Bc * SP * EP * 2);
  bf16*  attnb = (bf16*) take((size_t)Bc * SP * SP * 2);
  bf16*  n0b   = (bf16*) take((size_t)Bc * Dc * 2);
  bf16*  n1b   = (bf16*) take((size_t)Bc * Hc * 2);
  float* gpart = (float*)take((size_t)KS * Bc * Hc * 4);
  float* n2t   = (float*)take(256);

  // one-time prep (per call; deterministic)
  k_cvt_w1 <<<90000, 256, 0, stream>>>(W1, W1b);             // 92.16M elems / 4
  k_prep_x <<<12160, 256, 0, stream>>>(x, xb, xTb);
  k_prep_w0<<<380,   256, 0, stream>>>(W0, W0b);
  k_zero8  <<<1520,  256, 0, stream>>>(g0b);                 // zero incl. padding rows
  k_init_n0<<<11250, 256, 0, stream>>>(n0i, n0o, n0b);
  k_init_n12<<<32,   256, 0, stream>>>(n1i, n2i, n1o, n2o, n1b);
  k_wx     <<<1444,  256, 0, stream>>>(xb, W0b, Wx);         // 608*19 tiles, 8 waves/block

  for (int t = 0; t < TSTEPS; ++t){
    k_g0      <<<2250, 160, 0, stream>>>(n1b, W1b, Wx, g0b);     // uses old n1
    k_g1      <<<KS * 4, 256, 0, stream>>>(n0b, W1b, gpart);     // uses old n0
    k_g2      <<<1, 32, 0, stream>>>(n1o, W2, b2, n2t);          // uses old n1
    k_g1red   <<<32, 256, 0, stream>>>(gpart, b1, n2o, W2, n1o, n1b); // uses old n2, writes n1
    k_n2commit<<<1, 32, 0, stream>>>(n2t, n2o);                  // now commit new n2
    k_attn    <<<dim3(38, 16), 256, 0, stream>>>(g0b, xb, attnb);
    k_n0      <<<1444, 256, 0, stream>>>(attnb, xTb, n0o, n0b);  // writes new n0
  }
}
